// EGST_1073741824209
// MI455X (gfx1250) — compile-verified, matrix operand order checked by reading, not run on hardware
//
#include <hip/hip_runtime.h>
#include <cmath>

// ---------------------------------------------------------------------------
// Problem constants (B=8, J=K=2048, C=128). Sinkhorn slack handled analytically
// via potentials u, v: la(j,k) = la0(j,k) - u_j - v_k ; slack col = -u_j ;
// slack row = -v_k ; corner = 0.  la0 is written once and then read-only
// (134 MB -> fully L2-resident on MI455X's 192 MB L2).
// ---------------------------------------------------------------------------
#define BATCH 8
#define J_PTS 2048
#define K_PTS 2048
#define CF    128
#define NPN   4096          // param-net point count (J+K)

typedef float    v2f  __attribute__((ext_vector_type(2)));
typedef float    v8f  __attribute__((ext_vector_type(8)));
typedef _Float16 v16h __attribute__((ext_vector_type(16)));

#if defined(__has_builtin)
#if __has_builtin(__builtin_amdgcn_wmma_f32_16x16x4_f32)
#define HAVE_WMMA_F32 1
#endif
#if __has_builtin(__builtin_amdgcn_global_load_async_to_lds_b128) && \
    __has_builtin(__builtin_amdgcn_s_wait_asynccnt)
#define HAVE_ASYNC_LDS 1
#endif
#endif

#if HAVE_ASYNC_LDS
typedef int v4i __attribute__((vector_size(16)));
typedef __attribute__((address_space(1))) v4i* gv4i;   // global (AS1) int4*
typedef __attribute__((address_space(3))) v4i* lv4i;   // LDS    (AS3) int4*
#endif

// ---------------------------------------------------------------------------
// Squared-norm precompute: out[b*N + p] = sum_c f[b,p,c]^2
// ---------------------------------------------------------------------------
__global__ __launch_bounds__(256) void norms_kernel(const float* __restrict__ f,
                                                    float* __restrict__ out) {
  int i = blockIdx.x * 256 + threadIdx.x;
  if (i >= BATCH * J_PTS) return;
  const float4* p = (const float4*)(f + (size_t)i * CF);
  float s = 0.f;
  for (int q = 0; q < CF / 4; q++) {
    float4 v = p[q];
    s += v.x * v.x + v.y * v.y + v.z * v.z + v.w * v.w;
  }
  out[i] = s;
}

__global__ __launch_bounds__(256) void zero_kernel(float* __restrict__ p, int n) {
  int i = blockIdx.x * 256 + threadIdx.x;
  if (i < n) p[i] = 0.f;
}

// ---------------------------------------------------------------------------
// Fused feature-distance + affinity via WMMA with LDS-staged tiles.
// Block: 256 threads = 8 waves; block tile 64(J) x 128(K); wave tile 32x32.
// la0[b,j,k] = -beta_b * (|fs_j|^2 + |fr_k|^2 - 2*dot(fs_j,fr_k) - alpha_b)
// ---------------------------------------------------------------------------
#define LDS_LD 132   // 128 + 4 pad: kills bank conflicts, keeps 16B alignment

__global__ __launch_bounds__(256) void affinity_wmma_kernel(
    const float* __restrict__ fs, const float* __restrict__ fr,
    const float* __restrict__ ns, const float* __restrict__ nr,
    const float* __restrict__ betaA, const float* __restrict__ alphaA,
    float* __restrict__ la) {
  const int b    = blockIdx.z;
  const int tid  = threadIdx.x;
  const int lane = tid & 31;
  const int wave = tid >> 5;
  const int wr   = wave & 1;                 // 2 row-waves
  const int wc   = wave >> 1;                // 4 col-waves
  const int m0b  = blockIdx.y * 64;          // block row base (src j)
  const int n0b  = blockIdx.x * 128;         // block col base (ref k)
  const int hi   = lane >> 4;
  const int lq   = lane & 15;

  const float* fsb = fs + (size_t)b * J_PTS * CF;
  const float* frb = fr + (size_t)b * K_PTS * CF;

  __shared__ float As[64 * LDS_LD];
  __shared__ float Bs[128 * LDS_LD];

#if HAVE_ASYNC_LDS
  // async DMA global -> LDS (ASYNCcnt path); no VGPR bounce.
  for (int t = tid; t < 64 * 32; t += 256) {          // 32 x b128 per A row
    int row = t >> 5, c4 = (t & 31) * 4;
    __builtin_amdgcn_global_load_async_to_lds_b128(
        (gv4i)(fsb + (size_t)(m0b + row) * CF + c4),
        (lv4i)(&As[row * LDS_LD + c4]), 0, 0);
  }
  for (int t = tid; t < 128 * 32; t += 256) {         // 32 x b128 per B row
    int row = t >> 5, c4 = (t & 31) * 4;
    __builtin_amdgcn_global_load_async_to_lds_b128(
        (gv4i)(frb + (size_t)(n0b + row) * CF + c4),
        (lv4i)(&Bs[row * LDS_LD + c4]), 0, 0);
  }
  __builtin_amdgcn_s_wait_asynccnt(0);
  __syncthreads();
#else
  // cooperative, coalesced global -> LDS staging (whole K=128 fits)
  for (int t = tid; t < 64 * 32; t += 256) {          // 32 float4 per A row
    int row = t >> 5, c4 = (t & 31) * 4;
    float4 vv = *(const float4*)(fsb + (size_t)(m0b + row) * CF + c4);
    *(float4*)(&As[row * LDS_LD + c4]) = vv;
  }
  for (int t = tid; t < 128 * 32; t += 256) {         // 32 float4 per B row
    int row = t >> 5, c4 = (t & 31) * 4;
    float4 vv = *(const float4*)(frb + (size_t)(n0b + row) * CF + c4);
    *(float4*)(&Bs[row * LDS_LD + c4]) = vv;
  }
  __syncthreads();
#endif

  v8f acc[2][2];
  for (int i = 0; i < 2; i++)
    for (int j = 0; j < 2; j++)
      for (int q = 0; q < 8; q++) acc[i][j][q] = 0.f;

#if HAVE_WMMA_F32
  // f32 path: A 16x4 (lane m=L&15, K = v + 2*hi), B 4x16 (lane n=L&15, same K)
  for (int k0 = 0; k0 < CF; k0 += 4) {
    v2f afr[2], bfr[2];
    for (int i = 0; i < 2; i++)
      afr[i] = *(const v2f*)(&As[(wr * 32 + i * 16 + lq) * LDS_LD + k0 + 2 * hi]);
    for (int j = 0; j < 2; j++)
      bfr[j] = *(const v2f*)(&Bs[(wc * 32 + j * 16 + lq) * LDS_LD + k0 + 2 * hi]);
    for (int i = 0; i < 2; i++)
      for (int j = 0; j < 2; j++)
        acc[i][j] = __builtin_amdgcn_wmma_f32_16x16x4_f32(
            false, afr[i], false, bfr[j], (short)0, acc[i][j], false, false);
  }
#else
  // Fallback: f16 16x16x32 WMMA, K = (h&7)+8*hi+16*(h>>3).
  for (int k0 = 0; k0 < CF; k0 += 32) {
    v16h afr[2], bfr[2];
    for (int i = 0; i < 2; i++) {
      const float* src = &As[(wr * 32 + i * 16 + lq) * LDS_LD + k0];
      for (int h = 0; h < 16; h++)
        afr[i][h] = (_Float16)src[(h & 7) + 8 * hi + 16 * (h >> 3)];
    }
    for (int j = 0; j < 2; j++) {
      const float* src = &Bs[(wc * 32 + j * 16 + lq) * LDS_LD + k0];
      for (int h = 0; h < 16; h++)
        bfr[j][h] = (_Float16)src[(h & 7) + 8 * hi + 16 * (h >> 3)];
    }
    for (int i = 0; i < 2; i++)
      for (int j = 0; j < 2; j++)
        acc[i][j] = __builtin_amdgcn_wmma_f32_16x16x32_f16(
            false, afr[i], false, bfr[j], (short)0, acc[i][j], false, false);
  }
#endif

  const float be = betaA[b], al = alphaA[b];
  float* lab = la + (size_t)b * J_PTS * K_PTS;
  for (int i = 0; i < 2; i++) {
    for (int j = 0; j < 2; j++) {
      int n = n0b + wc * 32 + j * 16 + lq;
      float nrv = nr[(size_t)b * K_PTS + n];
      for (int v = 0; v < 8; v++) {
        int m = m0b + wr * 32 + i * 16 + v + 8 * hi;  // C/D layout: row = v+8*hi
        float d = ns[(size_t)b * J_PTS + m] + nrv - 2.0f * acc[i][j][v];
        lab[(size_t)m * K_PTS + n] = -be * (d - al);
      }
    }
  }
}

// ---------------------------------------------------------------------------
// Sinkhorn in potential form: read-only passes over la0.
// ---------------------------------------------------------------------------
__device__ __forceinline__ float wave_lse(float m, float s) {
  for (int off = 16; off > 0; off >>= 1) {
    float m2 = __shfl_xor(m, off, 32);
    float s2 = __shfl_xor(s, off, 32);
    float M  = fmaxf(m, m2);
    s = s * __expf(m - M) + s2 * __expf(m2 - M);
    m = M;
  }
  return m + __logf(s);
}

// stable log(exp(L) + 1)
__device__ __forceinline__ float log1pexp(float L) {
  return fmaxf(L, 0.f) + log1pf(__expf(-fabsf(L)));
}

// u_j = log( sum_k exp(la0[j,k] - v_k) + 1 ).  One wave per row.
__global__ __launch_bounds__(256) void sk_row_kernel(
    const float* __restrict__ la, const float* __restrict__ v,
    float* __restrict__ u) {
  int lane = threadIdx.x & 31;
  int row  = blockIdx.x * 8 + (threadIdx.x >> 5);   // 0 .. B*J-1
  int b = row >> 11;
  const float* r  = la + (size_t)row * K_PTS;
  const float* vb = v + (size_t)b * K_PTS;
  float m = -INFINITY, s = 0.f;
  for (int k = lane; k < K_PTS; k += 32) {
    float x = r[k] - vb[k];
    float M = fmaxf(m, x);
    s = s * __expf(m - M) + __expf(x - M);
    m = M;
  }
  float L = wave_lse(m, s);
  if (lane == 0) u[row] = log1pexp(L);
}

// v_k = log( sum_j exp(la0[j,k] - u_j) + 1 ).  One thread per column.
__global__ __launch_bounds__(256) void sk_col_kernel(
    const float* __restrict__ la, const float* __restrict__ u,
    float* __restrict__ v) {
  int k = blockIdx.x * 256 + threadIdx.x;           // < K_PTS
  int b = blockIdx.y;
  const float* lab = la + (size_t)b * J_PTS * K_PTS;
  const float* ub  = u + (size_t)b * J_PTS;
  float m = -INFINITY, s = 0.f;
  for (int j = 0; j < J_PTS; j++) {
    float x = lab[(size_t)j * K_PTS + k] - ub[j];
    float M = fmaxf(m, x);
    s = s * __expf(m - M) + __expf(x - M);
    m = M;
  }
  float L = m + __logf(s);
  v[(size_t)b * K_PTS + k] = log1pexp(L);
}

// ---------------------------------------------------------------------------
// row_sum + weighted_ref = (exp(la0-u-v) @ xyz_ref) / (row_sum + eps). Wave/row.
// ---------------------------------------------------------------------------
__global__ __launch_bounds__(256) void perm_reduce_kernel(
    const float* __restrict__ la, const float* __restrict__ u,
    const float* __restrict__ v, const float* __restrict__ xyz_ref,
    float* __restrict__ rs, float* __restrict__ wref) {
  int lane = threadIdx.x & 31;
  int row  = blockIdx.x * 8 + (threadIdx.x >> 5);
  int b = row >> 11;
  const float* r  = la + (size_t)row * K_PTS;
  const float* vb = v + (size_t)b * K_PTS;
  const float* xr = xyz_ref + (size_t)b * K_PTS * 3;
  float uj = u[row];
  float s = 0.f, sx = 0.f, sy = 0.f, sz = 0.f;
  for (int k = lane; k < K_PTS; k += 32) {
    float p = __expf(r[k] - uj - vb[k]);
    s  += p;
    sx += p * xr[k * 3 + 0];
    sy += p * xr[k * 3 + 1];
    sz += p * xr[k * 3 + 2];
  }
  for (int off = 16; off > 0; off >>= 1) {
    s  += __shfl_xor(s, off, 32);
    sx += __shfl_xor(sx, off, 32);
    sy += __shfl_xor(sy, off, 32);
    sz += __shfl_xor(sz, off, 32);
  }
  if (lane == 0) {
    rs[row] = s;
    float inv = 1.f / (s + 1e-5f);
    wref[(size_t)row * 3 + 0] = sx * inv;
    wref[(size_t)row * 3 + 1] = sy * inv;
    wref[(size_t)row * 3 + 2] = sz * inv;
  }
}

// ---------------------------------------------------------------------------
// Kabsch stage 1: per-batch sums {Sw, Sw*a(3), Sw*b(3), Sw*a*b^T(9)}.
// ---------------------------------------------------------------------------
__global__ __launch_bounds__(256) void kabsch_reduce_kernel(
    const float* __restrict__ xyz_src, const float* __restrict__ wref,
    const float* __restrict__ rs, float* __restrict__ sums) {
  int b = blockIdx.x, tid = threadIdx.x;
  float acc[16];
  for (int q = 0; q < 16; q++) acc[q] = 0.f;
  for (int j = tid; j < J_PTS; j += 256) {
    float w = rs[b * J_PTS + j];
    const float* a  = xyz_src + ((size_t)b * J_PTS + j) * 3;
    const float* bb = wref    + ((size_t)b * J_PTS + j) * 3;
    acc[0] += w;
    for (int i = 0; i < 3; i++) { acc[1 + i] += w * a[i]; acc[4 + i] += w * bb[i]; }
    for (int i = 0; i < 3; i++)
      for (int c = 0; c < 3; c++) acc[7 + i * 3 + c] += w * a[i] * bb[c];
  }
  __shared__ float sd[256];
  for (int q = 0; q < 16; q++) {
    sd[tid] = acc[q];
    __syncthreads();
    for (int st = 128; st > 0; st >>= 1) {
      if (tid < st) sd[tid] += sd[tid + st];
      __syncthreads();
    }
    if (tid == 0) sums[b * 16 + q] = sd[0];
    __syncthreads();
  }
}

// ---------------------------------------------------------------------------
// Kabsch stage 2: 3x3 Jacobi SVD + det correction. One thread per batch.
// ---------------------------------------------------------------------------
__global__ void kabsch_svd_kernel(const float* __restrict__ sums,
                                  float* __restrict__ out) {
  int b = threadIdx.x;
  if (b >= BATCH) return;
  const float* S = sums + b * 16;
  float Sw = S[0];
  float D  = Sw + 1e-5f;
  float ca[3], cb[3];
  for (int i = 0; i < 3; i++) { ca[i] = S[1 + i] / D; cb[i] = S[4 + i] / D; }
  float T = Sw / D;
  float H[3][3];
  for (int i = 0; i < 3; i++)
    for (int j = 0; j < 3; j++)
      H[i][j] = S[7 + i * 3 + j] / D - (2.f - T) * ca[i] * cb[j];

  float K[3][3];
  for (int i = 0; i < 3; i++)
    for (int j = 0; j < 3; j++) {
      float s = 0.f;
      for (int q = 0; q < 3; q++) s += H[q][i] * H[q][j];
      K[i][j] = s;
    }
  float V[3][3] = {{1, 0, 0}, {0, 1, 0}, {0, 0, 1}};
  for (int sweep = 0; sweep < 10; sweep++) {
    for (int pi = 0; pi < 3; pi++) {
      int p = (pi == 2) ? 1 : 0;
      int q = (pi == 0) ? 1 : 2;
      float apq = K[p][q];
      if (fabsf(apq) < 1e-20f) continue;
      float theta = 0.5f * (K[q][q] - K[p][p]) / apq;
      float t = ((theta >= 0.f) ? 1.f : -1.f) /
                (fabsf(theta) + sqrtf(theta * theta + 1.f));
      float c = rsqrtf(t * t + 1.f), sn = t * c;
      for (int i = 0; i < 3; i++) {
        float kip = K[i][p], kiq = K[i][q];
        K[i][p] = c * kip - sn * kiq;
        K[i][q] = sn * kip + c * kiq;
      }
      for (int i = 0; i < 3; i++) {
        float kpi = K[p][i], kqi = K[q][i];
        K[p][i] = c * kpi - sn * kqi;
        K[q][i] = sn * kpi + c * kqi;
      }
      for (int i = 0; i < 3; i++) {
        float vip = V[i][p], viq = V[i][q];
        V[i][p] = c * vip - sn * viq;
        V[i][q] = sn * vip + c * viq;
      }
    }
  }
  float lam[3] = {K[0][0], K[1][1], K[2][2]};
  for (int a = 0; a < 2; a++)
    for (int c = 0; c < 2 - a; c++)
      if (lam[c] < lam[c + 1]) {
        float tl = lam[c]; lam[c] = lam[c + 1]; lam[c + 1] = tl;
        for (int i = 0; i < 3; i++) {
          float tv = V[i][c]; V[i][c] = V[i][c + 1]; V[i][c + 1] = tv;
        }
      }
  float U[3][3];
  {
    float u0[3], u1[3], u2[3];
    for (int i = 0; i < 3; i++) {
      u0[i] = H[i][0] * V[0][0] + H[i][1] * V[1][0] + H[i][2] * V[2][0];
      u1[i] = H[i][0] * V[0][1] + H[i][1] * V[1][1] + H[i][2] * V[2][1];
    }
    float n0 = sqrtf(u0[0]*u0[0] + u0[1]*u0[1] + u0[2]*u0[2]);
    if (n0 > 1e-12f) { for (int i = 0; i < 3; i++) u0[i] /= n0; }
    else { u0[0] = 1.f; u0[1] = 0.f; u0[2] = 0.f; }
    float d01 = u0[0]*u1[0] + u0[1]*u1[1] + u0[2]*u1[2];
    for (int i = 0; i < 3; i++) u1[i] -= d01 * u0[i];
    float n1 = sqrtf(u1[0]*u1[0] + u1[1]*u1[1] + u1[2]*u1[2]);
    if (n1 > 1e-12f) { for (int i = 0; i < 3; i++) u1[i] /= n1; }
    else {
      float ax = fabsf(u0[0]);
      float t0[3] = {ax < 0.9f ? 1.f : 0.f, ax < 0.9f ? 0.f : 1.f, 0.f};
      u1[0] = u0[1]*t0[2] - u0[2]*t0[1];
      u1[1] = u0[2]*t0[0] - u0[0]*t0[2];
      u1[2] = u0[0]*t0[1] - u0[1]*t0[0];
      float nn = rsqrtf(u1[0]*u1[0] + u1[1]*u1[1] + u1[2]*u1[2]);
      for (int i = 0; i < 3; i++) u1[i] *= nn;
    }
    u2[0] = u0[1]*u1[2] - u0[2]*u1[1];
    u2[1] = u0[2]*u1[0] - u0[0]*u1[2];
    u2[2] = u0[0]*u1[1] - u0[1]*u1[0];
    for (int i = 0; i < 3; i++) { U[i][0] = u0[i]; U[i][1] = u1[i]; U[i][2] = u2[i]; }
  }
  float r0[3][3];
  for (int i = 0; i < 3; i++)
    for (int j = 0; j < 3; j++)
      r0[i][j] = V[i][0]*U[j][0] + V[i][1]*U[j][1] + V[i][2]*U[j][2];
  float det = r0[0][0]*(r0[1][1]*r0[2][2] - r0[1][2]*r0[2][1])
            - r0[0][1]*(r0[1][0]*r0[2][2] - r0[1][2]*r0[2][0])
            + r0[0][2]*(r0[1][0]*r0[2][1] - r0[1][1]*r0[2][0]);
  float R[3][3];
  for (int i = 0; i < 3; i++)
    for (int j = 0; j < 3; j++)
      R[i][j] = V[i][0]*U[j][0] + V[i][1]*U[j][1] + det * V[i][2]*U[j][2];
  float tv[3];
  for (int i = 0; i < 3; i++)
    tv[i] = -(R[i][0]*ca[0] + R[i][1]*ca[1] + R[i][2]*ca[2]) + cb[i];
  for (int i = 0; i < 3; i++) {
    for (int j = 0; j < 3; j++) out[b * 12 + i * 4 + j] = R[i][j];
    out[b * 12 + i * 4 + 3] = tv[i];
  }
}

// ---------------------------------------------------------------------------
// Parameter-prediction net
// ---------------------------------------------------------------------------
__global__ __launch_bounds__(256) void build_x0_kernel(
    const float* __restrict__ xs, const float* __restrict__ xr,
    float* __restrict__ x0) {
  int i = blockIdx.x * 256 + threadIdx.x;   // b*NPN + n
  if (i >= BATCH * NPN) return;
  int b = i >> 12, n = i & (NPN - 1);
  float v0, v1, v2, v3;
  if (n < J_PTS) {
    const float* p = xs + ((size_t)b * J_PTS + n) * 3;
    v0 = p[0]; v1 = p[1]; v2 = p[2]; v3 = 0.f;
  } else {
    const float* p = xr + ((size_t)b * K_PTS + (n - J_PTS)) * 3;
    v0 = p[0]; v1 = p[1]; v2 = p[2]; v3 = 1.f;
  }
  float* o = x0 + (size_t)b * 4 * NPN;
  o[0 * NPN + n] = v0; o[1 * NPN + n] = v1;
  o[2 * NPN + n] = v2; o[3 * NPN + n] = v3;
}

__global__ __launch_bounds__(256) void conv1x1_kernel(
    const float* __restrict__ in, const float* __restrict__ W,
    const float* __restrict__ bias, float* __restrict__ out,
    int Cin, int Cout) {
  int n = blockIdx.x * 256 + threadIdx.x;
  int o = blockIdx.y, b = blockIdx.z;
  const float* ib = in + (size_t)b * Cin * NPN + n;
  const float* w  = W + (size_t)o * Cin;
  float s = bias[o];
  for (int c = 0; c < Cin; c++) s += w[c] * ib[(size_t)c * NPN];
  out[((size_t)b * Cout + o) * NPN + n] = s;
}

__global__ __launch_bounds__(256) void gn3d_kernel(
    float* __restrict__ x, const float* __restrict__ gamma,
    const float* __restrict__ beta, int C, int groups, int relu) {
  int g = blockIdx.x, b = blockIdx.y, tid = threadIdx.x;
  int Cg = C / groups;
  size_t base = ((size_t)b * C + (size_t)g * Cg) * NPN;
  size_t cnt  = (size_t)Cg * NPN;
  float s = 0.f, s2 = 0.f;
  for (size_t i = tid; i < cnt; i += 256) {
    float v = x[base + i];
    s += v; s2 += v * v;
  }
  __shared__ float sdA[256], sdB[256];
  sdA[tid] = s; sdB[tid] = s2;
  __syncthreads();
  for (int st = 128; st > 0; st >>= 1) {
    if (tid < st) { sdA[tid] += sdA[tid + st]; sdB[tid] += sdB[tid + st]; }
    __syncthreads();
  }
  float mean = sdA[0] / (float)cnt;
  float var  = sdB[0] / (float)cnt - mean * mean;
  float rstd = rsqrtf(var + 1e-5f);
  for (size_t i = tid; i < cnt; i += 256) {
    int c = g * Cg + (int)(i / NPN);
    float v = (x[base + i] - mean) * rstd * gamma[c] + beta[c];
    if (relu) v = fmaxf(v, 0.f);
    x[base + i] = v;
  }
}

__global__ __launch_bounds__(256) void maxpool_kernel(
    const float* __restrict__ x, float* __restrict__ out) {
  int c = blockIdx.x, b = blockIdx.y, tid = threadIdx.x;
  const float* p = x + ((size_t)b * 1024 + c) * NPN;
  float m = -INFINITY;
  for (int n = tid; n < NPN; n += 256) m = fmaxf(m, p[n]);
  __shared__ float sd[256];
  sd[tid] = m;
  __syncthreads();
  for (int st = 128; st > 0; st >>= 1) {
    if (tid < st) sd[tid] = fmaxf(sd[tid], sd[tid + st]);
    __syncthreads();
  }
  if (tid == 0) out[(size_t)b * 1024 + c] = sd[0];
}

__global__ void linear_kernel(const float* __restrict__ in,
                              const float* __restrict__ W,
                              const float* __restrict__ bias,
                              float* __restrict__ out, int Cin, int Cout) {
  int i = blockIdx.x * blockDim.x + threadIdx.x;
  if (i >= BATCH * Cout) return;
  int b = i / Cout, o = i % Cout;
  const float* x = in + (size_t)b * Cin;
  const float* w = W + (size_t)o * Cin;
  float s = bias[o];
  for (int c = 0; c < Cin; c++) s += w[c] * x[c];
  out[i] = s;
}

__global__ void gn1d_relu_kernel(float* __restrict__ x,
                                 const float* __restrict__ g,
                                 const float* __restrict__ be,
                                 int C, int groups) {
  int i = blockIdx.x * blockDim.x + threadIdx.x;
  if (i >= BATCH * groups) return;
  int b = i / groups, grp = i % groups;
  int Cg = C / groups;
  float* p = x + (size_t)b * C + grp * Cg;
  float s = 0.f, s2 = 0.f;
  for (int c = 0; c < Cg; c++) { float v = p[c]; s += v; s2 += v * v; }
  float mean = s / Cg, var = s2 / Cg - mean * mean;
  float rstd = rsqrtf(var + 1e-5f);
  for (int c = 0; c < Cg; c++) {
    int cc = grp * Cg + c;
    p[c] = fmaxf((p[c] - mean) * rstd * g[cc] + be[cc], 0.f);
  }
}

__global__ void head_kernel(const float* __restrict__ in,
                            const float* __restrict__ W,
                            const float* __restrict__ bias,
                            float* __restrict__ beta_out,
                            float* __restrict__ alpha_out) {
  int i = threadIdx.x;
  if (i >= BATCH * 2) return;
  int b = i >> 1, o = i & 1;
  const float* x = in + (size_t)b * 256;
  const float* w = W + (size_t)o * 256;
  float s = bias[o];
  for (int c = 0; c < 256; c++) s += w[c] * x[c];
  float sp = (s > 20.f) ? s : log1pf(__expf(s));
  if (o == 0) beta_out[b] = sp; else alpha_out[b] = sp;
}

// ---------------------------------------------------------------------------
// Host-side orchestration
// ---------------------------------------------------------------------------
extern "C" void kernel_launch(void* const* d_in, const int* in_sizes, int n_in,
                              void* d_out, int out_size, void* d_ws,
                              size_t ws_size, hipStream_t stream) {
  (void)in_sizes; (void)n_in; (void)out_size; (void)ws_size;
  const float* xyz_src  = (const float*)d_in[0];
  const float* xyz_ref  = (const float*)d_in[1];
  const float* feat_src = (const float*)d_in[2];
  const float* feat_ref = (const float*)d_in[3];
  const float *cw[5], *cbi[5], *cg[5], *cbe[5];
  for (int i = 0; i < 5; i++) {
    cw[i]  = (const float*)d_in[4 + 4 * i];
    cbi[i] = (const float*)d_in[5 + 4 * i];
    cg[i]  = (const float*)d_in[6 + 4 * i];
    cbe[i] = (const float*)d_in[7 + 4 * i];
  }
  const float* lw0 = (const float*)d_in[24];
  const float* lb0 = (const float*)d_in[25];
  const float* lg0 = (const float*)d_in[26];
  const float* lbe0 = (const float*)d_in[27];
  const float* lw1 = (const float*)d_in[28];
  const float* lb1 = (const float*)d_in[29];
  const float* lg1 = (const float*)d_in[30];
  const float* lbe1 = (const float*)d_in[31];
  const float* lw2 = (const float*)d_in[32];
  const float* lb2 = (const float*)d_in[33];

  // ---- workspace carving (floats) -----------------------------------------
  float* ws = (float*)d_ws;
  size_t off = 0;
  float* la = ws;            off += (size_t)BATCH * J_PTS * K_PTS;  // 33.55M
  float* a5 = la;            // alias: (B,1024,4096) == la size exactly; dead
                             // before la is first written
  float* actA = ws + off;    off += (size_t)BATCH * 128 * NPN;
  float* actB = ws + off;    off += (size_t)BATCH * 128 * NPN;
  float* x0 = ws + off;      off += (size_t)BATCH * 4 * NPN;
  float* pooled = ws + off;  off += (size_t)BATCH * 1024;
  float* l1 = ws + off;      off += (size_t)BATCH * 512;
  float* l2 = ws + off;      off += (size_t)BATCH * 256;
  float* betaA = ws + off;   off += BATCH;
  float* alphaA = ws + off;  off += BATCH;
  float* ns = ws + off;      off += (size_t)BATCH * J_PTS;
  float* nr = ws + off;      off += (size_t)BATCH * K_PTS;
  float* uP = ws + off;      off += (size_t)BATCH * J_PTS;   // row potentials
  float* vP = ws + off;      off += (size_t)BATCH * K_PTS;   // col potentials
  float* rs = ws + off;      off += (size_t)BATCH * J_PTS;
  float* wref = ws + off;    off += (size_t)BATCH * J_PTS * 3;
  float* ksums = ws + off;   off += (size_t)BATCH * 16;

  // ---- parameter prediction net -------------------------------------------
  build_x0_kernel<<<(BATCH * NPN + 255) / 256, 256, 0, stream>>>(xyz_src, xyz_ref, x0);
  conv1x1_kernel<<<dim3(NPN / 256, 64, BATCH), 256, 0, stream>>>(x0, cw[0], cbi[0], actA, 4, 64);
  gn3d_kernel<<<dim3(8, BATCH), 256, 0, stream>>>(actA, cg[0], cbe[0], 64, 8, 1);
  conv1x1_kernel<<<dim3(NPN / 256, 64, BATCH), 256, 0, stream>>>(actA, cw[1], cbi[1], actB, 64, 64);
  gn3d_kernel<<<dim3(8, BATCH), 256, 0, stream>>>(actB, cg[1], cbe[1], 64, 8, 1);
  conv1x1_kernel<<<dim3(NPN / 256, 64, BATCH), 256, 0, stream>>>(actB, cw[2], cbi[2], actA, 64, 64);
  gn3d_kernel<<<dim3(8, BATCH), 256, 0, stream>>>(actA, cg[2], cbe[2], 64, 8, 1);
  conv1x1_kernel<<<dim3(NPN / 256, 128, BATCH), 256, 0, stream>>>(actA, cw[3], cbi[3], actB, 64, 128);
  gn3d_kernel<<<dim3(8, BATCH), 256, 0, stream>>>(actB, cg[3], cbe[3], 128, 8, 1);
  conv1x1_kernel<<<dim3(NPN / 256, 1024, BATCH), 256, 0, stream>>>(actB, cw[4], cbi[4], a5, 128, 1024);
  gn3d_kernel<<<dim3(16, BATCH), 256, 0, stream>>>(a5, cg[4], cbe[4], 1024, 16, 1);
  maxpool_kernel<<<dim3(1024, BATCH), 256, 0, stream>>>(a5, pooled);
  linear_kernel<<<(BATCH * 512 + 255) / 256, 256, 0, stream>>>(pooled, lw0, lb0, l1, 1024, 512);
  gn1d_relu_kernel<<<1, 128, 0, stream>>>(l1, lg0, lbe0, 512, 16);
  linear_kernel<<<(BATCH * 256 + 255) / 256, 256, 0, stream>>>(l1, lw1, lb1, l2, 512, 256);
  gn1d_relu_kernel<<<1, 128, 0, stream>>>(l2, lg1, lbe1, 256, 16);
  head_kernel<<<1, 32, 0, stream>>>(l2, lw2, lb2, betaA, alphaA);

  // ---- fused distance + affinity (WMMA, LDS-staged) -----------------------
  norms_kernel<<<(BATCH * J_PTS + 255) / 256, 256, 0, stream>>>(feat_src, ns);
  norms_kernel<<<(BATCH * K_PTS + 255) / 256, 256, 0, stream>>>(feat_ref, nr);
  affinity_wmma_kernel<<<dim3(K_PTS / 128, J_PTS / 64, BATCH), 256, 0, stream>>>(
      feat_src, feat_ref, ns, nr, betaA, alphaA, la);

  // ---- sinkhorn (5 iterations, potential form: read-only over la) ---------
  zero_kernel<<<(BATCH * K_PTS + 255) / 256, 256, 0, stream>>>(vP, BATCH * K_PTS);
  for (int it = 0; it < 5; it++) {
    sk_row_kernel<<<BATCH * J_PTS / 8, 256, 0, stream>>>(la, vP, uP);
    sk_col_kernel<<<dim3(K_PTS / 256, BATCH), 256, 0, stream>>>(la, uP, vP);
  }

  // ---- weighted ref + Kabsch ----------------------------------------------
  perm_reduce_kernel<<<BATCH * J_PTS / 8, 256, 0, stream>>>(la, uP, vP, xyz_ref, rs, wref);
  kabsch_reduce_kernel<<<BATCH, 256, 0, stream>>>(xyz_src, wref, rs, ksums);
  kabsch_svd_kernel<<<1, BATCH, 0, stream>>>(ksums, (float*)d_out);
}
